// MultiheadSelfAttention_43576738185906
// MI455X (gfx1250) — matrix operand order repaired and checked by reading, not compiled, not run on hardware
//
#include <hip/hip_runtime.h>
#include <hip/hip_bf16.h>

typedef __bf16 bf16;
typedef __attribute__((ext_vector_type(16))) __bf16 v16bf;
typedef __attribute__((ext_vector_type(8)))  __bf16 v8bf;
typedef __attribute__((ext_vector_type(8)))  float  v8f;

#define T_LEN 4096
#define NBATCH 2
#define EMBED 1024
#define HEADS 16
#define HEAD_DIM 64
#define WIN 512
#define MROWS (T_LEN * NBATCH)   // 8192, row m = t*NBATCH + n

// ---------------- fragment loaders ----------------
// A operand (16xK, 16-bit): lane holds row M = lane%16.
// element e -> K = klo + e            for e in [0,8)   (klo = 8*(lane>=16))
//              K = klo + 16 + (e-8)   for e in [8,16)
static __device__ inline v16bf ld_pair(const bf16* p0, const bf16* p1) {
    v8bf a = *(const v8bf*)p0;
    v8bf b = *(const v8bf*)p1;
    return __builtin_shufflevector(a, b, 0,1,2,3,4,5,6,7,8,9,10,11,12,13,14,15);
}
static __device__ inline v16bf ldA(const bf16* row, int klo) {
    return ld_pair(row + klo, row + klo + 16);
}
// B operand (Kx16, 16-bit): lane holds column N = lane%16.
// element e -> K = koff + e (e<8), koff + 16 + (e-8) (e>=8), koff = 8*(lane>=16)  (same as A)
static __device__ inline v16bf ldB(const bf16* col, int koff) {
    return ld_pair(col + koff, col + koff + 16);
}
static __device__ inline v8f wmma_bf16(v16bf a, v16bf b, v8f c) {
    return __builtin_amdgcn_wmma_f32_16x16x32_bf16(false, a, false, b, (short)0, c, false, false);
}

// ---------------- fp32 -> bf16 convert ----------------
__global__ void cvt_f32_bf16(const float* __restrict__ in, bf16* __restrict__ out, int n) {
    int i = blockIdx.x * blockDim.x + threadIdx.x;
    if (i < n) out[i] = (bf16)in[i];
}

// ---------------- GEMM: C[M x OUT] = A[M x 1024] * W[OUT x 1024]^T + bias ----------------
// block = 256 threads = 8 waves (4 row-groups x 2 col-groups); block tile 64 x 128;
// each wave computes a 16 x 64 tile as 4 WMMA accumulators.
// K loop is software-pipelined: next iteration's 5 fragments are in flight
// while the current iteration's 4 WMMAs execute.
template <bool OUT_BF16>
__global__ void gemm_bias(const bf16* __restrict__ A, const bf16* __restrict__ W,
                          const float* __restrict__ bias, void* __restrict__ outp, int OUT) {
    const int lane = threadIdx.x & 31;
    const int wave = threadIdx.x >> 5;
    const int wr = wave & 3, wc = wave >> 2;
    const int row0 = blockIdx.y * 64 + wr * 16;
    const int col0 = blockIdx.x * 128 + wc * 64;
    const int lm  = lane & 15;
    const int klo = (lane >> 4) << 3;   // A interleave
    const int kfb = (lane >> 4) << 3;   // B interleave (same as A)

    const bf16* arow = A + (size_t)(row0 + lm) * EMBED;
    const bf16* wrow[4];
#pragma unroll
    for (int j = 0; j < 4; ++j) wrow[j] = W + (size_t)(col0 + j * 16 + lm) * EMBED;

    v8f acc[4] = {};
    v16bf a = ldA(arow, klo);
    v16bf b[4];
#pragma unroll
    for (int j = 0; j < 4; ++j) b[j] = ldB(wrow[j], kfb);

    for (int k0 = 0; k0 < EMBED; k0 += 32) {
        const int kn = k0 + 32;
        v16bf an;
        v16bf bn[4];
        if (kn < EMBED) {
            an = ldA(arow + kn, klo);
#pragma unroll
            for (int j = 0; j < 4; ++j) bn[j] = ldB(wrow[j] + kn, kfb);
            __builtin_prefetch(arow + kn + 32, 0, 0);   // global_prefetch_b8, no counter
        }
#pragma unroll
        for (int j = 0; j < 4; ++j) acc[j] = wmma_bf16(a, b[j], acc[j]);
        a = an;
#pragma unroll
        for (int j = 0; j < 4; ++j) b[j] = bn[j];
    }

    const int hi = lane >> 4;
#pragma unroll
    for (int j = 0; j < 4; ++j) {
        const int col = col0 + j * 16 + lm;
        const float bv = bias[col];
#pragma unroll
        for (int r = 0; r < 8; ++r) {
            const int row = row0 + r + 8 * hi;
            float v = acc[j][r] + bv;
            if (OUT_BF16) ((bf16*)outp)[(size_t)row * OUT + col] = (bf16)v;
            else          ((float*)outp)[(size_t)row * OUT + col] = v;
        }
    }
}

// ---------------- RoPE + repack ----------------
// qkv: [MROWS x 3072] bf16 (row m = t*NBATCH + n), cols [0,1024)=q, [1024,2048)=k, [2048,3072)=v
// q,k out: [n][h][t][d]; v out transposed: [n][h][d][t]
__global__ void rope_repack(const bf16* __restrict__ qkv, bf16* __restrict__ qo,
                            bf16* __restrict__ ko, bf16* __restrict__ vt) {
    int idx = blockIdx.x * blockDim.x + threadIdx.x;   // N*T*H*32
    const int i = idx & 31;          // rotary pair index
    int rest = idx >> 5;
    const int h = rest & 15;  rest >>= 4;
    const int t = rest & 4095;
    const int n = rest >> 12;

    const float inv = __powf(10000.0f, -(float)i * (1.0f / 32.0f));
    const float ang = (float)(t + WIN) * inv;
    const float c = cosf(ang), s = sinf(ang);

    const size_t m = (size_t)t * NBATCH + n;
    const bf16* base = qkv + m * 3072;
    const size_t ho = ((size_t)(n * HEADS + h));

    // q
    {
        float x1 = (float)base[h * 64 + i];
        float x2 = (float)base[h * 64 + 32 + i];
        bf16* q = qo + (ho * T_LEN + t) * HEAD_DIM;
        q[i]      = (bf16)(x1 * c - x2 * s);
        q[i + 32] = (bf16)(x1 * s + x2 * c);
    }
    // k
    {
        float x1 = (float)base[1024 + h * 64 + i];
        float x2 = (float)base[1024 + h * 64 + 32 + i];
        bf16* k = ko + (ho * T_LEN + t) * HEAD_DIM;
        k[i]      = (bf16)(x1 * c - x2 * s);
        k[i + 32] = (bf16)(x1 * s + x2 * c);
    }
    // v (transpose: [d][t])
    {
        bf16* v = vt + ho * HEAD_DIM * T_LEN;
        v[(size_t)i * T_LEN + t]        = base[2048 + h * 64 + i];
        v[(size_t)(i + 32) * T_LEN + t] = base[2048 + h * 64 + 32 + i];
    }
}

// ---------------- flash-style sliding-window attention ----------------
// one wave handles one (n, h, 16-query tile); 128 threads = 4 waves per block.
// Masked/scaled scores overwrite the WMMA accumulator registers in place to
// keep VGPR pressure below the spill threshold.
__global__ void attn_kernel(const bf16* __restrict__ Q, const bf16* __restrict__ K,
                            const bf16* __restrict__ VT, bf16* __restrict__ O) {
    __shared__ bf16 plds[4][16 * 32];
    const int lane = threadIdx.x & 31;
    const int wave = threadIdx.x >> 5;
    const int task = blockIdx.x * 4 + wave;     // N*H*(T/16) = 8192 tasks
    const int qtile = task & 255;
    const int h = (task >> 8) & 15;
    const int n = task >> 12;
    const int qbase = qtile * 16;
    const int lm = lane & 15, hi = lane >> 4;
    const int klo = hi << 3;
    const int kfb = hi << 3;

    const size_t ho = (size_t)(n * HEADS + h);
    const bf16* qh = Q  + ho * T_LEN * HEAD_DIM;
    const bf16* kh = K  + ho * T_LEN * HEAD_DIM;
    const bf16* vh = VT + ho * HEAD_DIM * T_LEN;

    const bf16* qrow = qh + (size_t)(qbase + lm) * HEAD_DIM;
    v16bf qa0 = ldA(qrow, klo);
    v16bf qa1 = ldA(qrow + 32, klo);

    v8f o[4] = {};
    float mrow[8], lrow[8];
#pragma unroll
    for (int r = 0; r < 8; ++r) { mrow[r] = -1e30f; lrow[r] = 0.0f; }

    int ks0 = qbase - WIN; if (ks0 < 0) ks0 = 0; ks0 &= ~31;
    const float scale = 0.125f;   // 1/sqrt(64)

    for (int kstart = ks0; kstart <= qbase + 15; kstart += 32) {
        // S = Q * K^T for this 16x32 key chunk (two 16x16 C fragments)
        v8f s[2] = {};
#pragma unroll
        for (int j = 0; j < 2; ++j) {
            const bf16* krow = kh + (size_t)(kstart + j * 16 + lm) * HEAD_DIM;
            s[j] = wmma_bf16(qa0, ldB(krow, kfb), s[j]);
            s[j] = wmma_bf16(qa1, ldB(krow + 32, kfb), s[j]);
        }
        // mask + scale IN PLACE, per-row max
        float mx[8];
#pragma unroll
        for (int r = 0; r < 8; ++r) mx[r] = -1e30f;
#pragma unroll
        for (int j = 0; j < 2; ++j)
#pragma unroll
            for (int r = 0; r < 8; ++r) {
                const int row = r + 8 * hi;
                const int qa_abs = qbase + row;
                const int ka = kstart + j * 16 + lm;
                const bool valid = (ka <= qa_abs) && (ka + WIN >= qa_abs);
                float v = valid ? s[j][r] * scale : -1e30f;
                s[j][r] = v;
                mx[r] = fmaxf(mx[r], v);
            }
#pragma unroll
        for (int off = 1; off < 16; off <<= 1)
#pragma unroll
            for (int r = 0; r < 8; ++r)
                mx[r] = fmaxf(mx[r], __shfl_xor(mx[r], off, 32));
        float alpha[8], psum[8];
#pragma unroll
        for (int r = 0; r < 8; ++r) {
            float mn = fmaxf(mrow[r], mx[r]);
            alpha[r] = __expf(mrow[r] - mn);
            mrow[r] = mn;
            psum[r] = 0.0f;
        }
        // P = exp(S - m), stage to LDS for C-layout -> A-layout transpose
#pragma unroll
        for (int j = 0; j < 2; ++j)
#pragma unroll
            for (int r = 0; r < 8; ++r) {
                float p = (s[j][r] > -1e29f) ? __expf(s[j][r] - mrow[r]) : 0.0f;
                psum[r] += p;
                plds[wave][(r + 8 * hi) * 32 + j * 16 + lm] = (bf16)p;
            }
#pragma unroll
        for (int off = 1; off < 16; off <<= 1)
#pragma unroll
            for (int r = 0; r < 8; ++r)
                psum[r] += __shfl_xor(psum[r], off, 32);
#pragma unroll
        for (int r = 0; r < 8; ++r) lrow[r] = lrow[r] * alpha[r] + psum[r];
#pragma unroll
        for (int t4 = 0; t4 < 4; ++t4)
#pragma unroll
            for (int r = 0; r < 8; ++r) o[t4][r] *= alpha[r];

        // O += P * V   (P: 16x32 A operand from LDS, V: 32 keys x 64 dims)
        v16bf pa = ldA(&plds[wave][lm * 32], klo);
#pragma unroll
        for (int t4 = 0; t4 < 4; ++t4) {
            const bf16* vcol = vh + (size_t)(t4 * 16 + lm) * T_LEN + kstart;
            o[t4] = wmma_bf16(pa, ldB(vcol, kfb), o[t4]);
        }
    }

    // normalize and store to attn buffer [MROWS x 1024], row m = t*NBATCH + n
#pragma unroll
    for (int t4 = 0; t4 < 4; ++t4)
#pragma unroll
        for (int r = 0; r < 8; ++r) {
            const int row = r + 8 * hi;
            const int trow = qbase + row;
            float v = o[t4][r] / lrow[r];
            O[((size_t)trow * NBATCH + n) * EMBED + h * 64 + t4 * 16 + lm] = (bf16)v;
        }
}

// ---------------- host orchestration ----------------
extern "C" void kernel_launch(void* const* d_in, const int* in_sizes, int n_in,
                              void* d_out, int out_size, void* d_ws, size_t ws_size,
                              hipStream_t stream) {
    (void)in_sizes; (void)n_in; (void)out_size; (void)ws_size;
    const float* x     = (const float*)d_in[0];
    const float* w_q   = (const float*)d_in[1];
    const float* b_q   = (const float*)d_in[2];
    const float* w_kv  = (const float*)d_in[3];
    const float* b_kv  = (const float*)d_in[4];
    const float* w_out = (const float*)d_in[5];
    const float* b_out = (const float*)d_in[6];

    char* ws = (char*)d_ws;
    size_t off = 0;
    auto take = [&](size_t bytes) { char* p = ws + off; off = (off + bytes + 255) & ~(size_t)255; return p; };

    bf16*  xbf   = (bf16*)take((size_t)MROWS * EMBED * 2);       // x as [8192 x 1024] (t,n) rows
    bf16*  wqkv  = (bf16*)take((size_t)3072 * EMBED * 2);        // w_q rows then w_kv rows
    bf16*  woutb = (bf16*)take((size_t)EMBED * EMBED * 2);
    bf16*  qkv   = (bf16*)take((size_t)MROWS * 3072 * 2);
    bf16*  qr    = (bf16*)take((size_t)NBATCH * HEADS * T_LEN * HEAD_DIM * 2);
    bf16*  kr    = (bf16*)take((size_t)NBATCH * HEADS * T_LEN * HEAD_DIM * 2);
    bf16*  vt    = (bf16*)take((size_t)NBATCH * HEADS * HEAD_DIM * T_LEN * 2);
    bf16*  attn  = (bf16*)take((size_t)MROWS * EMBED * 2);
    float* bqkv  = (float*)take((size_t)3072 * 4);

    // 1) converts (x is (T,B,C) which is already row m = t*NBATCH+n major)
    {
        int n = MROWS * EMBED;
        cvt_f32_bf16<<<(n + 255) / 256, 256, 0, stream>>>(x, xbf, n);
        n = EMBED * EMBED;
        cvt_f32_bf16<<<(n + 255) / 256, 256, 0, stream>>>(w_q, wqkv, n);
        n = 2 * EMBED * EMBED;
        cvt_f32_bf16<<<(n + 255) / 256, 256, 0, stream>>>(w_kv, wqkv + (size_t)EMBED * EMBED, n);
        n = EMBED * EMBED;
        cvt_f32_bf16<<<(n + 255) / 256, 256, 0, stream>>>(w_out, woutb, n);
    }
    hipMemcpyAsync(bqkv, b_q, 1024 * sizeof(float), hipMemcpyDeviceToDevice, stream);
    hipMemcpyAsync(bqkv + 1024, b_kv, 2048 * sizeof(float), hipMemcpyDeviceToDevice, stream);

    // 2) fused QKV projection: [8192 x 3072]
    {
        dim3 grid(3072 / 128, MROWS / 64);
        gemm_bias<true><<<grid, 256, 0, stream>>>(xbf, wqkv, bqkv, (void*)qkv, 3072);
    }
    // 3) RoPE + repack
    {
        int n = NBATCH * T_LEN * HEADS * 32;
        rope_repack<<<n / 256, 256, 0, stream>>>(qkv, qr, kr, vt);
    }
    // 4) sliding-window attention: 8192 wave tasks, 4 waves per block
    attn_kernel<<<2048, 128, 0, stream>>>(qr, kr, vt, attn);

    // 5) output projection straight into d_out (fp32, (T,B,C) row-major = m-major)
    {
        dim3 grid(EMBED / 128, MROWS / 64);
        gemm_bias<false><<<grid, 256, 0, stream>>>(attn, woutb, b_out, d_out, EMBED);
    }
}
